// DMGNN_10264971837836
// MI455X (gfx1250) — matrix operand order repaired and checked by reading, not compiled, not run on hardware
//
#include <hip/hip_runtime.h>
#include <hip/hip_bf16.h>

// =====================================================================
// DMGNN forward for MI455X (gfx1250, wave32).
//
// Dense 64/128-dim GEMMs -> v_wmma_f32_16x16x32_f16 (f16 in, f32 acc),
// templated on DIN so the K-loop fully unrolls (2 or 4 WMMAs per tile,
// loads in one s_clause of global_load_b128).
// Sparse edge ops (GIN agg, SDDMM, edge softmax, SpMM, pooling) ->
// edge-parallel kernels with f32 global atomics; node tables (12.8 MB)
// are L2-resident on the 192 MB L2, so scattered traffic never touches HBM.
//
// Param flattening assumption: inputs appear in setup_inputs() insertion
// order, params recursively in dict-insertion order:
//   [0]=h [1]=edge_src [2]=edge_dst [3]=edge_weight [4]=graph_ids
//   fil{Wn,Ws}x3, gin{W0,b0,g0,b0n,W1,b1,ga,ba,go,bo_}x4,
//   gt{Wq,bq,Wk,bk,Wv,bv,Wo,bo,g1,b1,Wf1,bf1,Wf2,bf2,g2,b2}x5,
//   pred{W,b}x5, pred_ts{W,b}x5.
// =====================================================================

typedef __attribute__((ext_vector_type(16))) _Float16 v16h;
typedef __attribute__((ext_vector_type(8)))  float    v8f;

#define HID   64
#define HEADS 4
#define HD    16
#define OUTD  16
#define NGR   64
#define BN_EPS 1e-5f

// ---------------------------------------------------------------------
// Generic zero / add
// ---------------------------------------------------------------------
__global__ void zero_kernel(float* __restrict__ p, long long n) {
    long long i = (long long)blockIdx.x * blockDim.x + threadIdx.x;
    long long stride = (long long)gridDim.x * blockDim.x;
    for (; i < n; i += stride) p[i] = 0.0f;
}

__global__ void add_kernel(const float* __restrict__ a, const float* __restrict__ b,
                           float* __restrict__ y, int n) {
    int i = blockIdx.x * blockDim.x + threadIdx.x;
    if (i < n) y[i] = a[i] + b[i];
}

// ---------------------------------------------------------------------
// WMMA GEMM:  Y[N x dout] = epi( (X[N x DIN] @ W^T) + bias, resid, scale, relu )
//   W row-major (dout x DIN).  epi: y = (acc+bias)*scale [+resid] [relu]
// Block = 128 threads = 4 waves; wave w owns the 16x16 tile at
// (blockIdx.x*16, blockIdx.y*64 + w*16).  DIN in {64,128}, dout mult of 64.
//
// Fragment layouts (CDNA5 ISA 7.12.2):
//   A (16x32 f16): lane l: m=l&15, h=l>>4; elem e -> K = kb + 8h + e (e<8)
//                  or kb + 16 + 8h + (e-8).   (two contiguous float4 pairs)
//   B (32x16 f16): lane l: n=l&15, h=l>>4; elem e -> K = same as A (kb + 8h + e, or kb + 16 + 8h + (e-8)).
//   C/D (16x16 f32): lane l: n=l&15; vgpr r -> m = r + 8*(l>>4).
//
// Row guard: A rows are CLAMPED (not zeroed) -- row m of A only feeds
// row m of C, and out-of-range rows are masked at the store, so clamped
// garbage is harmless.  This keeps loads unconditional & vectorizable
// (no exec-mask churn) and EXEC all-ones for the WMMAs.
// ---------------------------------------------------------------------
template <int DIN>
__launch_bounds__(128)
__global__ void gemm_wmma_kernel(const float* __restrict__ X, const float* __restrict__ W,
                                 const float* __restrict__ bias, const float* __restrict__ resid,
                                 float* __restrict__ Y, int N, int dout,
                                 float scale, int relu) {
    const int lane = threadIdx.x & 31;
    const int wave = threadIdx.x >> 5;
    const int rowBase = blockIdx.x * 16;
    const int colBase = blockIdx.y * 64 + wave * 16;
    const int idx16 = lane & 15;
    const int half  = lane >> 4;

    int row = rowBase + idx16;
    if (row >= N) row = N - 1;            // clamp (see note above)
    const int col = colBase + idx16;
    const float* __restrict__ xr = X + (size_t)row * DIN;
    const float* __restrict__ wr = W + (size_t)col * DIN;

    v8f acc = {};
    #pragma unroll
    for (int kb = 0; kb < DIN; kb += 32) {
        v16h a, b;
        // ---- A fragment: 4 x float4 (16B aligned: offsets mult. of 8 floats)
        const float4 a0 = *reinterpret_cast<const float4*>(xr + kb + 8 * half);
        const float4 a1 = *reinterpret_cast<const float4*>(xr + kb + 8 * half + 4);
        const float4 a2 = *reinterpret_cast<const float4*>(xr + kb + 16 + 8 * half);
        const float4 a3 = *reinterpret_cast<const float4*>(xr + kb + 16 + 8 * half + 4);
        a[0]=(_Float16)a0.x;  a[1]=(_Float16)a0.y;  a[2]=(_Float16)a0.z;  a[3]=(_Float16)a0.w;
        a[4]=(_Float16)a1.x;  a[5]=(_Float16)a1.y;  a[6]=(_Float16)a1.z;  a[7]=(_Float16)a1.w;
        a[8]=(_Float16)a2.x;  a[9]=(_Float16)a2.y;  a[10]=(_Float16)a2.z; a[11]=(_Float16)a2.w;
        a[12]=(_Float16)a3.x; a[13]=(_Float16)a3.y; a[14]=(_Float16)a3.z; a[15]=(_Float16)a3.w;
        // ---- B fragment: same K order as A (two contiguous float4 pairs)
        const float4 b0 = *reinterpret_cast<const float4*>(wr + kb + 8 * half);
        const float4 b1 = *reinterpret_cast<const float4*>(wr + kb + 8 * half + 4);
        const float4 b2 = *reinterpret_cast<const float4*>(wr + kb + 16 + 8 * half);
        const float4 b3 = *reinterpret_cast<const float4*>(wr + kb + 16 + 8 * half + 4);
        b[0]=(_Float16)b0.x;  b[1]=(_Float16)b0.y;  b[2]=(_Float16)b0.z;  b[3]=(_Float16)b0.w;
        b[4]=(_Float16)b1.x;  b[5]=(_Float16)b1.y;  b[6]=(_Float16)b1.z;  b[7]=(_Float16)b1.w;
        b[8]=(_Float16)b2.x;  b[9]=(_Float16)b2.y;  b[10]=(_Float16)b2.z; b[11]=(_Float16)b2.w;
        b[12]=(_Float16)b3.x; b[13]=(_Float16)b3.y; b[14]=(_Float16)b3.z; b[15]=(_Float16)b3.w;

        acc = __builtin_amdgcn_wmma_f32_16x16x32_f16(
                  false, a, false, b, (short)0, acc, false, false);
    }

    const float bi = bias ? bias[col] : 0.0f;
    #pragma unroll
    for (int r = 0; r < 8; ++r) {
        const int m = rowBase + r + 8 * half;
        if (m < N) {
            float v = (acc[r] + bi) * scale;
            if (resid) v += resid[(size_t)m * dout + col];
            if (relu)  v = fmaxf(v, 0.0f);
            Y[(size_t)m * dout + col] = v;
        }
    }
}

// ---------------------------------------------------------------------
// Small dense GEMM for FIL layers (din in {1,32}): thread per output elem.
// ---------------------------------------------------------------------
__global__ void gemm_small_kernel(const float* __restrict__ X, const float* __restrict__ W,
                                  float* __restrict__ Y, int N, int din, int dout) {
    int i = blockIdx.x * blockDim.x + threadIdx.x;
    if (i >= N * dout) return;
    int nr = i / dout, o = i % dout;
    const float* xr = X + (size_t)nr * din;
    const float* wr = W + (size_t)o * din;
    float s = 0.0f;
    for (int k = 0; k < din; ++k) s += xr[k] * wr[k];
    Y[i] = s;
}

// ---------------------------------------------------------------------
// FIL: degree, u_mul_e message scatter, self+mean-neigh combine
// ---------------------------------------------------------------------
__global__ void deg_kernel(const int* __restrict__ dst, float* __restrict__ deg, int E) {
    int e = blockIdx.x * blockDim.x + threadIdx.x;
    if (e < E) atomicAdd(&deg[dst[e]], 1.0f);
}

__global__ void fil_msg_kernel(const float* __restrict__ HS, const int* __restrict__ src,
                               const int* __restrict__ dst, const float* __restrict__ w,
                               float* __restrict__ NEI, int E, int F) {
    int i = blockIdx.x * blockDim.x + threadIdx.x;
    if (i >= E * F) return;
    int e = i / F, f = i % F;
    atomicAdd(&NEI[(size_t)dst[e] * F + f], HS[(size_t)src[e] * F + f] * w[e]);
}

__global__ void fil_combine_kernel(const float* __restrict__ X, const float* __restrict__ Ws,
                                   const float* __restrict__ NEI, const float* __restrict__ deg,
                                   float* __restrict__ Y, int N, int din, int dout, int relu) {
    int i = blockIdx.x * blockDim.x + threadIdx.x;
    if (i >= N * dout) return;
    int nr = i / dout, o = i % dout;
    const float* xr = X + (size_t)nr * din;
    const float* wr = Ws + (size_t)o * din;
    float s = 0.0f;
    for (int k = 0; k < din; ++k) s += xr[k] * wr[k];
    float v = s + NEI[i] / fmaxf(deg[nr], 1.0f);
    Y[i] = relu ? fmaxf(v, 0.0f) : v;
}

// ---------------------------------------------------------------------
// GIN sum aggregation (64 features)
// ---------------------------------------------------------------------
__global__ void gin_agg_kernel(const float* __restrict__ X, const int* __restrict__ src,
                               const int* __restrict__ dst, float* __restrict__ AGG, int E) {
    int i = blockIdx.x * blockDim.x + threadIdx.x;
    if (i >= E * HID) return;
    int e = i >> 6, f = i & 63;
    atomicAdd(&AGG[(size_t)dst[e] * HID + f], X[(size_t)src[e] * HID + f]);
}

// ---------------------------------------------------------------------
// BatchNorm: block-reduced partial (sum, sumsq) -> stats[0..63]=sum,
// stats[64..127]=sumsq -> apply.
// ---------------------------------------------------------------------
__global__ void bn_stats_kernel(const float* __restrict__ X, float* __restrict__ stats, int N) {
    __shared__ float ssum[256], ssq[256];
    const int f = threadIdx.x & 63;        // feature
    const int g = threadIdx.x >> 6;        // 0..3 row group
    float s = 0.0f, q = 0.0f;
    for (int r = blockIdx.x * 4 + g; r < N; r += gridDim.x * 4) {
        float v = X[(size_t)r * HID + f];
        s += v; q += v * v;
    }
    ssum[threadIdx.x] = s; ssq[threadIdx.x] = q;
    __syncthreads();
    if (g == 0) {
        for (int j = 1; j < 4; ++j) { s += ssum[j * 64 + f]; q += ssq[j * 64 + f]; }
        atomicAdd(&stats[f], s);
        atomicAdd(&stats[64 + f], q);
    }
}

__global__ void bn_apply_kernel(const float* __restrict__ X, const float* __restrict__ stats,
                                const float* __restrict__ gam, const float* __restrict__ bet,
                                float* __restrict__ Y, int N, float invN, int relu) {
    int i = blockIdx.x * blockDim.x + threadIdx.x;
    if (i >= N * HID) return;
    int f = i & 63;
    float mean = stats[f] * invN;
    float var  = stats[64 + f] * invN - mean * mean;
    float v = gam[f] * (X[i] - mean) * rsqrtf(var + BN_EPS) + bet[f];
    Y[i] = relu ? fmaxf(v, 0.0f) : v;
}

// ---------------------------------------------------------------------
// MHA edge ops.  Q/K/V are N x 64, flat index j = d*HEADS + h.
// Segment-max via order-preserving uint encoding + atomicMax.
// ---------------------------------------------------------------------
__device__ __forceinline__ unsigned f2mono(float f) {
    unsigned u = __float_as_uint(f);
    return (u & 0x80000000u) ? ~u : (u | 0x80000000u);
}
__device__ __forceinline__ float mono2f(unsigned u) {
    return (u & 0x80000000u) ? __uint_as_float(u ^ 0x80000000u) : __uint_as_float(~u);
}

__global__ void edge_score_kernel(const float* __restrict__ Q, const float* __restrict__ K,
                                  const int* __restrict__ src, const int* __restrict__ dst,
                                  const float* __restrict__ w, float* __restrict__ escore,
                                  unsigned* __restrict__ mEnc, int E) {
    int e = blockIdx.x * blockDim.x + threadIdx.x;
    if (e >= E) return;
    int s_ = src[e], d_ = dst[e];
    const float* q = Q + (size_t)s_ * HID;
    const float* k = K + (size_t)d_ * HID;
    float acc[HEADS] = {0.f, 0.f, 0.f, 0.f};
    #pragma unroll
    for (int j = 0; j < HID; ++j) acc[j & 3] += q[j] * k[j];
    const float we = w[e];
    #pragma unroll
    for (int h = 0; h < HEADS; ++h) {
        float sc = acc[h] * we;
        escore[(size_t)e * HEADS + h] = sc;
        atomicMax(&mEnc[s_ * HEADS + h], f2mono(sc));
    }
}

__global__ void edge_expsum_kernel(const float* __restrict__ escore, const unsigned* __restrict__ mEnc,
                                   const int* __restrict__ src, float* __restrict__ eexp,
                                   float* __restrict__ ssum, int E) {
    int i = blockIdx.x * blockDim.x + threadIdx.x;
    if (i >= E * HEADS) return;
    int e = i >> 2, h = i & 3;
    float m = mono2f(mEnc[src[e] * HEADS + h]);
    float ex = __expf(escore[i] - m);
    eexp[i] = ex;
    atomicAdd(&ssum[src[e] * HEADS + h], ex);
}

__global__ void edge_norm_kernel(float* __restrict__ eexp, const float* __restrict__ ssum,
                                 const int* __restrict__ src, int E) {
    int i = blockIdx.x * blockDim.x + threadIdx.x;
    if (i >= E * HEADS) return;
    int e = i >> 2, h = i & 3;
    eexp[i] /= ssum[src[e] * HEADS + h];
}

__global__ void edge_aggregate_kernel(const float* __restrict__ alpha, const float* __restrict__ V,
                                      const int* __restrict__ src, const int* __restrict__ dst,
                                      float* __restrict__ O, int E) {
    int i = blockIdx.x * blockDim.x + threadIdx.x;
    if (i >= E * HID) return;
    int e = i >> 6, j = i & 63;       // j = d*HEADS + h
    int h = j & 3;
    float a = alpha[(size_t)e * HEADS + h];
    atomicAdd(&O[(size_t)src[e] * HID + j], a * V[(size_t)dst[e] * HID + j]);
}

// ---------------------------------------------------------------------
// Graph sum-pooling + per-layer linear head  (score += pool @ W^T + b)
// ---------------------------------------------------------------------
__global__ void pool_kernel(const float* __restrict__ X, const int* __restrict__ gid,
                            float* __restrict__ pool, int N) {
    int i = blockIdx.x * blockDim.x + threadIdx.x;
    if (i >= N * HID) return;
    int nr = i >> 6, f = i & 63;
    atomicAdd(&pool[gid[nr] * HID + f], X[i]);
}

__global__ void head_kernel(const float* __restrict__ pool, const float* __restrict__ W,
                            const float* __restrict__ b, float* __restrict__ score) {
    int i = blockIdx.x * blockDim.x + threadIdx.x;
    if (i >= NGR * OUTD) return;
    int g = i / OUTD, o = i % OUTD;
    const float* pr = pool + (size_t)g * HID;
    const float* wr = W + (size_t)o * HID;
    float s = b[o];
    for (int k = 0; k < HID; ++k) s += pr[k] * wr[k];
    score[i] += s;   // sequential launches, one writer per element
}

// =====================================================================
// Host orchestration
// =====================================================================
extern "C" void kernel_launch(void* const* d_in, const int* in_sizes, int n_in,
                              void* d_out, int out_size, void* d_ws, size_t ws_size,
                              hipStream_t stream) {
    const float* h    = (const float*)d_in[0];
    const int*   esrc = (const int*)d_in[1];
    const int*   edst = (const int*)d_in[2];
    const float* ew   = (const float*)d_in[3];
    const int*   gid  = (const int*)d_in[4];
    const int N = in_sizes[0];
    const int E = in_sizes[1];

    // ---- params (insertion-order flattening; see header comment) ----
    const float* P[256];
    for (int i = 0; i < n_in && i < 256; ++i) P[i] = (const float*)d_in[i];
    int t = 5;
    const float *filWn[3], *filWs[3];
    for (int i = 0; i < 3; ++i) { filWn[i] = P[t++]; filWs[i] = P[t++]; }
    const float *gW0[4], *gb0[4], *gg0[4], *gb0n[4], *gW1[4], *gb1[4],
                *gga[4], *gba[4], *ggo[4], *gbo[4];
    for (int i = 0; i < 4; ++i) {
        gW0[i]=P[t++]; gb0[i]=P[t++]; gg0[i]=P[t++]; gb0n[i]=P[t++]; gW1[i]=P[t++];
        gb1[i]=P[t++]; gga[i]=P[t++]; gba[i]=P[t++]; ggo[i]=P[t++]; gbo[i]=P[t++];
    }
    const float *tWq[5], *tbq[5], *tWk[5], *tbk[5], *tWv[5], *tbv[5], *tWo[5], *tbo[5],
                *tg1[5], *tb1[5], *tWf1[5], *tbf1[5], *tWf2[5], *tbf2[5], *tg2[5], *tb2[5];
    for (int j = 0; j < 5; ++j) {
        tWq[j]=P[t++]; tbq[j]=P[t++]; tWk[j]=P[t++]; tbk[j]=P[t++];
        tWv[j]=P[t++]; tbv[j]=P[t++]; tWo[j]=P[t++]; tbo[j]=P[t++];
        tg1[j]=P[t++]; tb1[j]=P[t++]; tWf1[j]=P[t++]; tbf1[j]=P[t++];
        tWf2[j]=P[t++]; tbf2[j]=P[t++]; tg2[j]=P[t++]; tb2[j]=P[t++];
    }
    const float *pW[5], *pB[5], *qW[5], *qB[5];
    for (int i = 0; i < 5; ++i) { pW[i] = P[t++]; pB[i] = P[t++]; }
    for (int i = 0; i < 5; ++i) { qW[i] = P[t++]; qB[i] = P[t++]; }

    // ---- workspace carve ----
    char* base = (char*)d_ws;
    auto alloc = [&](size_t nfloats) -> float* {
        float* p = (float*)base;
        base += ((nfloats * sizeof(float) + 255) / 256) * 256;
        return p;
    };
    float* deg  = alloc(N);
    float* x    = alloc((size_t)N * HID);     // GIN stream (h0 first)
    float* ts   = alloc((size_t)N * HID);     // GT-layer output
    float* qb   = alloc((size_t)N * HID);
    float* kb   = alloc((size_t)N * HID);
    float* vb   = alloc((size_t)N * HID);
    float* t0   = alloc((size_t)N * HID);
    float* t1   = alloc((size_t)N * HID);
    size_t bigN = (size_t)N * 128;
    size_t bigE = (size_t)E * 2 * HEADS;
    float* big  = alloc(bigN > bigE ? bigN : bigE);  // FFN temp AND edge score/exp (disjoint in time)
    float* ffn    = big;
    float* escore = big;
    float* eexp   = big + (size_t)E * HEADS;
    unsigned* mEnc = (unsigned*)alloc((size_t)N * HEADS);
    float* ssum = alloc((size_t)N * HEADS);
    float* stats = alloc(256);
    float* pool  = alloc((size_t)NGR * HID);
    float* score = (float*)d_out;

    auto cdiv = [](long long a, long long b) -> unsigned { return (unsigned)((a + b - 1) / b); };
    auto zero = [&](float* p, long long n) {
        zero_kernel<<<1024, 256, 0, stream>>>(p, n);
    };
    auto gemm = [&](const float* X, const float* W, const float* bias, const float* resid,
                    float* Y, int din, int dout, float scale, int relu) {
        dim3 grid(cdiv(N, 16), dout / 64);
        if (din == 64)
            gemm_wmma_kernel<64><<<grid, 128, 0, stream>>>(X, W, bias, resid, Y, N, dout, scale, relu);
        else
            gemm_wmma_kernel<128><<<grid, 128, 0, stream>>>(X, W, bias, resid, Y, N, dout, scale, relu);
    };
    auto bn = [&](const float* X, const float* gg, const float* bb, float* Y, int relu) {
        zero(stats, 128);
        bn_stats_kernel<<<256, 256, 0, stream>>>(X, stats, N);
        bn_apply_kernel<<<cdiv((long long)N * HID, 256), 256, 0, stream>>>(
            X, stats, gg, bb, Y, N, 1.0f / (float)N, relu);
    };
    auto poolHead = [&](const float* X, const float* W, const float* B) {
        zero(pool, NGR * HID);
        pool_kernel<<<cdiv((long long)N * HID, 256), 256, 0, stream>>>(X, gid, pool, N);
        head_kernel<<<cdiv(NGR * OUTD, 256), 256, 0, stream>>>(pool, W, B, score);
    };
    auto gtLayer = [&](const float* hin, int j, float* out) {
        const float scaling = 0.25f;   // HD^-0.5 = 16^-0.5
        gemm(hin, tWq[j], tbq[j], nullptr, qb, HID, HID, scaling, 0);
        gemm(hin, tWk[j], tbk[j], nullptr, kb, HID, HID, 1.0f, 0);
        gemm(hin, tWv[j], tbv[j], nullptr, vb, HID, HID, 1.0f, 0);
        zero((float*)mEnc, (long long)N * HEADS);       // 0 == monotonic(-inf floor)
        zero(ssum, (long long)N * HEADS);
        zero(t0, (long long)N * HID);                   // attention output accumulator
        edge_score_kernel<<<cdiv(E, 256), 256, 0, stream>>>(qb, kb, esrc, edst, ew, escore, mEnc, E);
        edge_expsum_kernel<<<cdiv((long long)E * HEADS, 256), 256, 0, stream>>>(escore, mEnc, esrc, eexp, ssum, E);
        edge_norm_kernel<<<cdiv((long long)E * HEADS, 256), 256, 0, stream>>>(eexp, ssum, esrc, E);
        edge_aggregate_kernel<<<cdiv((long long)E * HID, 256), 256, 0, stream>>>(eexp, vb, esrc, edst, t0, E);
        gemm(t0, tWo[j], tbo[j], hin, t1, HID, HID, 1.0f, 0);   // Wo proj + residual h1
        bn(t1, tg1[j], tb1[j], t1, 0);                          // t1 = h2
        gemm(t1, tWf1[j], tbf1[j], nullptr, ffn, HID, 128, 1.0f, 1);
        gemm(ffn, tWf2[j], tbf2[j], t1, t0, 128, HID, 1.0f, 0); // + residual h2
        bn(t0, tg2[j], tb2[j], out, 0);
    };

    // ================= pipeline =================
    zero(score, (long long)NGR * OUTD);

    // ---- FILmodule: 3 FILConv layers ----
    zero(deg, N);
    deg_kernel<<<cdiv(E, 256), 256, 0, stream>>>(edst, deg, E);
    {
        const float* cur = h;
        int din = 1;
        const int douts[3] = {32, 32, HID};
        for (int i = 0; i < 3; ++i) {
            const int dout = douts[i];
            gemm_small_kernel<<<cdiv((long long)N * dout, 256), 256, 0, stream>>>(
                cur, filWn[i], qb, N, din, dout);                       // hs = x @ Wn^T
            zero(kb, (long long)N * dout);
            fil_msg_kernel<<<cdiv((long long)E * dout, 256), 256, 0, stream>>>(
                qb, esrc, edst, ew, kb, E, dout);                       // scatter hs[src]*w
            float* dst_ = (i == 2) ? x : ((i == 0) ? t0 : t1);
            fil_combine_kernel<<<cdiv((long long)N * dout, 256), 256, 0, stream>>>(
                cur, filWs[i], kb, deg, dst_, N, din, dout, (i < 2) ? 1 : 0);
            cur = dst_;
            din = dout;
        }
    }
    // x == h0

    poolHead(x, pW[0], pB[0]);
    gtLayer(x, 0, ts);
    poolHead(ts, qW[0], qB[0]);

    for (int i = 0; i < 4; ++i) {
        // ---- GIN layer ----
        zero(t0, (long long)N * HID);
        gin_agg_kernel<<<cdiv((long long)E * HID, 256), 256, 0, stream>>>(x, esrc, edst, t0, E);
        add_kernel<<<cdiv((long long)N * HID, 256), 256, 0, stream>>>(x, t0, t1, N * HID);
        gemm(t1, gW0[i], gb0[i], nullptr, t0, HID, HID, 1.0f, 0);
        bn(t0, gg0[i], gb0n[i], t0, 1);
        gemm(t0, gW1[i], gb1[i], nullptr, t1, HID, HID, 1.0f, 0);
        bn(t1, gga[i], gba[i], t1, 1);    // ApplyNodeFunc BN+ReLU
        bn(t1, ggo[i], gbo[i], x, 1);     // DMGNN outer BN+ReLU
        poolHead(x, pW[i + 1], pB[i + 1]);
        // ---- GT layer ----
        gtLayer(x, i + 1, ts);
        poolHead(ts, qW[i + 1], qB[i + 1]);
    }
    (void)out_size; (void)ws_size; (void)n_in;
}
